// Dgm_17282948399612
// MI455X (gfx1250) — compile-verified, matrix operand order checked by reading, not run on hardware
//
#include <hip/hip_runtime.h>
#include <hip/hip_bf16.h>
#include <stdint.h>

// Problem constants (match reference)
#define B_  8192
#define D_  784
#define H_  256
#define L_  64
#define T_  10
#define DP  800          // D padded to a multiple of 32 (WMMA K granularity)
#define MROWS 32         // M-tile rows per block (2 x 16-row WMMA subtiles)
#define MAXT_TILES 256   // ceil(B/MROWS): worst case all rows in one task bucket

typedef __bf16 v16bf __attribute__((ext_vector_type(16)));
typedef float  v8f   __attribute__((ext_vector_type(8)));

// Async global->LDS path (CDNA5 GLOBAL_LOAD_ASYNC_TO_LDS_*, ASYNCcnt),
// guarded so the kernel still builds on toolchains without the builtins.
#if defined(__has_builtin)
#  if __has_builtin(__builtin_amdgcn_global_load_async_to_lds_b128) && \
      __has_builtin(__builtin_amdgcn_s_wait_asynccnt)
#    define USE_ASYNC_LDS 1
#  endif
#endif
#ifndef USE_ASYNC_LDS
#  define USE_ASYNC_LDS 0
#endif

#if USE_ASYNC_LDS
// Parameter types per hipcc diagnostic: int4 pointers in AS1 (global) / AS3 (LDS)
typedef int v4i_ __attribute__((vector_size(16)));
typedef __attribute__((address_space(1))) v4i_* g_v4i_ptr;
typedef __attribute__((address_space(3))) v4i_* l_v4i_ptr;
#endif

static __device__ __forceinline__ uint16_t f2bf(float f) {
  union { float f; uint32_t u; } v; v.f = f;
  uint32_t r = v.u + 0x7FFFu + ((v.u >> 16) & 1u);   // round-to-nearest-even
  return (uint16_t)(r >> 16);
}

// ---------------- bucketing kernels ----------------
__global__ void k_zero_i32(int* p, int n) {
  int i = blockIdx.x * blockDim.x + threadIdx.x;
  if (i < n) p[i] = 0;
}
__global__ void k_count(const int* __restrict__ task, int* __restrict__ counts) {
  int b = blockIdx.x * blockDim.x + threadIdx.x;
  if (b < B_) atomicAdd(&counts[task[b]], 1);
}
__global__ void k_scan(const int* __restrict__ counts, int* __restrict__ offsets,
                       int* __restrict__ cursors) {
  if (threadIdx.x == 0 && blockIdx.x == 0) {
    int a = 0;
    for (int t = 0; t < T_; ++t) { offsets[t] = a; cursors[t] = a; a += counts[t]; }
    offsets[T_] = a;
  }
}
__global__ void k_scatter(const int* __restrict__ task, int* __restrict__ cursors,
                          int* __restrict__ rowidx) {
  int b = blockIdx.x * blockDim.x + threadIdx.x;
  if (b < B_) {
    int p = atomicAdd(&cursors[task[b]], 1);
    rowidx[p] = b;
  }
}

// ---------------- conversion / sampling kernels ----------------
__global__ void k_cvt_x(const float* __restrict__ x, uint16_t* __restrict__ xb) {
  int i = blockIdx.x * blockDim.x + threadIdx.x;
  if (i >= B_ * DP) return;
  int r = i / DP, c = i - r * DP;
  xb[i] = (c < D_) ? f2bf(x[(size_t)r * D_ + c]) : (uint16_t)0;
}

// W: Tn x K x N (f32, row-major)  ->  Wt: Tn x N x Kp (bf16), zero-padded K
__global__ void k_cvt_wt(const float* __restrict__ W, uint16_t* __restrict__ Wt,
                         int Tn, int K, int N, int Kp) {
  int i = blockIdx.x * blockDim.x + threadIdx.x;
  int total = Tn * N * Kp;
  if (i >= total) return;
  int t = i / (N * Kp);
  int rem = i - t * (N * Kp);
  int n = rem / Kp, k = rem - n * Kp;
  float v = 0.f;
  if (k < K) v = W[((size_t)t * K + k) * N + n];
  Wt[i] = f2bf(v);
}

// Bayesian sample + transpose + convert: Wt[t][n][k] = bf16(mu + exp(ls)*eps)
__global__ void k_sample_wt(const float* __restrict__ wmu, const float* __restrict__ wls,
                            const float* __restrict__ ew, uint16_t* __restrict__ Wt,
                            int Tn, int K, int N, int Kp) {
  int i = blockIdx.x * blockDim.x + threadIdx.x;
  int total = Tn * N * Kp;
  if (i >= total) return;
  int t = i / (N * Kp);
  int rem = i - t * (N * Kp);
  int n = rem / Kp, k = rem - n * Kp;
  float v = 0.f;
  if (k < K) {
    size_t o = ((size_t)t * K + k) * N + n;
    v = wmu[o] + __expf(wls[o]) * ew[o];
  }
  Wt[i] = f2bf(v);
}

__global__ void k_sample_b(const float* __restrict__ bmu, const float* __restrict__ bls,
                           const float* __restrict__ eb, float* __restrict__ out, int n) {
  int i = blockIdx.x * blockDim.x + threadIdx.x;
  if (i < n) out[i] = bmu[i] + __expf(bls[i]) * eb[i];
}

// z = mu + exp(ls)*eps ; also emit mu / log_sigma to d_out
__global__ void k_z(const float* __restrict__ enc_out, const float* __restrict__ eps_z,
                    float* __restrict__ mu_out, float* __restrict__ ls_out,
                    uint16_t* __restrict__ zb) {
  int i = blockIdx.x * blockDim.x + threadIdx.x;
  if (i >= B_ * L_) return;
  int b = i / L_, k = i - b * L_;
  float mu = enc_out[(size_t)b * (2 * L_) + k];
  float ls = enc_out[(size_t)b * (2 * L_) + L_ + k];
  mu_out[i] = mu;
  ls_out[i] = ls;
  zb[i] = f2bf(mu + __expf(ls) * eps_z[i]);
}

// ---------------- WMMA tile GEMM ----------------
// 256 threads = 8 wave32s. Each block computes a 32-row tile of rows that all
// share one task (bucketed rowidx). Each wave owns 32 output columns and keeps
// 4 accumulators (2 M-subtiles x 2 N-subtiles); each B fragment is loaded once
// from the pre-transposed bf16 weights Wt[N][K] and reused for both M-subtiles
// (halves L2 weight traffic vs a 16-row tile). The 32 x K A-tile is staged in
// LDS, via GLOBAL_LOAD_ASYNC_TO_LDS_B128 when available.
// mode: 0 = relu -> bf16 out, 1 = identity -> f32 out, 2 = sigmoid -> f32 out
__global__ __launch_bounds__(256) void k_gemm(
    const uint16_t* __restrict__ A, const uint16_t* __restrict__ Wt,
    const float* __restrict__ bias,
    float* __restrict__ outF, uint16_t* __restrict__ outB,
    const int* __restrict__ rowidx, const int* __restrict__ offsets,
    int N, int K, int ldo, int mode) {
  __shared__ __align__(16) uint16_t ldsA[MROWS * DP];   // 50 KB max
  __shared__ int ldsRow[MROWS];

  int t = 0, tile, base, cnt;
  if (offsets) {
    t = blockIdx.x / MAXT_TILES;
    tile = blockIdx.x - t * MAXT_TILES;
    base = offsets[t];
    cnt = offsets[t + 1] - base;
    if (tile * MROWS >= cnt) return;   // uniform: whole block exits
  } else {
    tile = blockIdx.x; base = 0; cnt = B_;
  }
  const uint16_t* Wt_t   = Wt + (size_t)t * N * K;
  const float*    bias_t = bias + (size_t)t * N;

  // stage the 32 gathered row indices once
  if (threadIdx.x < MROWS) {
    int j = tile * MROWS + (int)threadIdx.x;
    int gr = -1;
    if (j < cnt) gr = offsets ? rowidx[base + j] : j;
    ldsRow[threadIdx.x] = gr;
  }
  __syncthreads();

  // cooperative fill of the 32 x K bf16 A-tile into LDS (b128 chunks)
  int q4 = K >> 3;                 // uint4 per row
  int total = MROWS * q4;
  for (int q = threadIdx.x; q < total; q += 256) {
    int r = q / q4, c = q - r * q4;
    int gr = ldsRow[r];
#if USE_ASYNC_LDS
    if (gr >= 0) {
      g_v4i_ptr gsrc = (g_v4i_ptr)(uintptr_t)(A + (size_t)gr * K + (size_t)c * 8);
      l_v4i_ptr ldst = (l_v4i_ptr)(uint32_t)(uintptr_t)
                           (ldsA + (size_t)r * K + (size_t)c * 8);
      __builtin_amdgcn_global_load_async_to_lds_b128(gsrc, ldst, 0, 0);
    } else {
      ((uint4*)ldsA)[q] = make_uint4(0u, 0u, 0u, 0u);
    }
#else
    uint4 v = make_uint4(0u, 0u, 0u, 0u);
    if (gr >= 0) v = ((const uint4*)(A + (size_t)gr * K))[c];
    ((uint4*)ldsA)[q] = v;
#endif
  }
#if USE_ASYNC_LDS
  __builtin_amdgcn_s_wait_asynccnt(0);
#endif
  __syncthreads();

  int wave = threadIdx.x >> 5;
  int lane = threadIdx.x & 31;
  int colBase = (blockIdx.y * 8 + wave) * 32;
  if (colBase >= N) return;

  int mrow  = lane & 15;           // A: lane half picks K 0-15 / 16-31
  int khalf = lane >> 4;

  int n0 = colBase + (lane & 15);  // B: lane = output column within 16
  int n1 = n0 + 16;
  int n0c = n0 < N ? n0 : N - 1;   // clamp; out-of-range columns masked at store
  int n1c = n1 < N ? n1 : N - 1;
  const uint16_t* bp0 = Wt_t + (size_t)n0c * K + khalf * 16;
  const uint16_t* bp1 = Wt_t + (size_t)n1c * K + khalf * 16;
  const uint16_t* ap0 = ldsA + mrow * K + khalf * 16;          // rows 0-15
  const uint16_t* ap1 = ldsA + (16 + mrow) * K + khalf * 16;   // rows 16-31

  union AB { uint4 q[2]; v16bf v; };
  v8f acc00 = {}; v8f acc01 = {}; v8f acc10 = {}; v8f acc11 = {};

  for (int kb = 0; kb < K; kb += 32) {
    AB a0, a1, b0, b1;
    a0.q[0] = ((const uint4*)(ap0 + kb))[0];
    a0.q[1] = ((const uint4*)(ap0 + kb))[1];
    a1.q[0] = ((const uint4*)(ap1 + kb))[0];
    a1.q[1] = ((const uint4*)(ap1 + kb))[1];
    b0.q[0] = ((const uint4*)(bp0 + kb))[0];
    b0.q[1] = ((const uint4*)(bp0 + kb))[1];
    b1.q[0] = ((const uint4*)(bp1 + kb))[0];
    b1.q[1] = ((const uint4*)(bp1 + kb))[1];
    __builtin_prefetch(bp0 + kb + 128, 0, 1);
    __builtin_prefetch(bp1 + kb + 128, 0, 1);
    acc00 = __builtin_amdgcn_wmma_f32_16x16x32_bf16(false, a0.v, false, b0.v,
                                                    (short)0, acc00, false, false);
    acc01 = __builtin_amdgcn_wmma_f32_16x16x32_bf16(false, a0.v, false, b1.v,
                                                    (short)0, acc01, false, false);
    acc10 = __builtin_amdgcn_wmma_f32_16x16x32_bf16(false, a1.v, false, b0.v,
                                                    (short)0, acc10, false, false);
    acc11 = __builtin_amdgcn_wmma_f32_16x16x32_bf16(false, a1.v, false, b1.v,
                                                    (short)0, acc11, false, false);
  }

  // C layout: lanes 0-15 hold M=0..7 (VGPR j -> M=j), lanes 16-31 hold M=8..15
  int mbase = khalf * 8;
  for (int msub = 0; msub < 2; ++msub) {
    for (int nsub = 0; nsub < 2; ++nsub) {
      int col = colBase + nsub * 16 + (lane & 15);
      if (col >= N) continue;
      union { v8f v; float f[8]; } u;
      u.v = msub ? (nsub ? acc11 : acc10) : (nsub ? acc01 : acc00);
      float bv = bias_t[col];
      for (int j = 0; j < 8; ++j) {
        int gr = ldsRow[msub * 16 + mbase + j];
        if (gr < 0) continue;
        float vv = u.f[j] + bv;
        if (mode == 0)      outB[(size_t)gr * ldo + col] = f2bf(fmaxf(vv, 0.f));
        else if (mode == 1) outF[(size_t)gr * ldo + col] = vv;
        else                outF[(size_t)gr * ldo + col] = 1.f / (1.f + __expf(-vv));
      }
    }
  }
}

// ---------------- host launcher ----------------
static inline int div_up(long a, long b) { return (int)((a + b - 1) / b); }

extern "C" void kernel_launch(void* const* d_in, const int* in_sizes, int n_in,
                              void* d_out, int out_size, void* d_ws, size_t ws_size,
                              hipStream_t stream) {
  (void)in_sizes; (void)n_in; (void)out_size; (void)ws_size;
  const float* x       = (const float*)d_in[0];
  const int*   task    = (const int*)  d_in[1];
  const float* enc_W1  = (const float*)d_in[2];
  const float* enc_b1  = (const float*)d_in[3];
  const float* enc_W2  = (const float*)d_in[4];
  const float* enc_b2  = (const float*)d_in[5];
  const float* enc_W3  = (const float*)d_in[6];
  const float* enc_b3  = (const float*)d_in[7];
  const float* enc_W4  = (const float*)d_in[8];
  const float* enc_b4  = (const float*)d_in[9];
  const float* s1_wmu  = (const float*)d_in[10];
  const float* s1_wls  = (const float*)d_in[11];
  const float* s1_bmu  = (const float*)d_in[12];
  const float* s1_bls  = (const float*)d_in[13];
  const float* s2_wmu  = (const float*)d_in[14];
  const float* s2_wls  = (const float*)d_in[15];
  const float* s2_bmu  = (const float*)d_in[16];
  const float* s2_bls  = (const float*)d_in[17];
  const float* h1_wmu  = (const float*)d_in[18];
  const float* h1_wls  = (const float*)d_in[19];
  const float* h1_bmu  = (const float*)d_in[20];
  const float* h1_bls  = (const float*)d_in[21];
  const float* h2_wmu  = (const float*)d_in[22];
  const float* h2_wls  = (const float*)d_in[23];
  const float* h2_bmu  = (const float*)d_in[24];
  const float* h2_bls  = (const float*)d_in[25];
  const float* eps_s1_w = (const float*)d_in[26];
  const float* eps_s1_b = (const float*)d_in[27];
  const float* eps_s2_w = (const float*)d_in[28];
  const float* eps_s2_b = (const float*)d_in[29];
  const float* eps_h1_w = (const float*)d_in[30];
  const float* eps_h1_b = (const float*)d_in[31];
  const float* eps_h2_w = (const float*)d_in[32];
  const float* eps_h2_b = (const float*)d_in[33];
  const float* eps_z    = (const float*)d_in[34];

  // d_out = gen (B x D) | mu (B x L) | log_sigma (B x L)
  float* genO = (float*)d_out;
  float* muO  = genO + (size_t)B_ * D_;
  float* lsO  = muO + (size_t)B_ * L_;

  // workspace carve-up (256B aligned)
  char* ws = (char*)d_ws;
  size_t off = 0;
  auto alloc = [&](size_t bytes) -> void* {
    off = (off + 255) & ~(size_t)255;
    void* p = ws + off;
    off += bytes;
    return p;
  };
  int* counts  = (int*)alloc(T_ * sizeof(int));
  int* offsets = (int*)alloc((T_ + 1) * sizeof(int));
  int* cursors = (int*)alloc(T_ * sizeof(int));
  int* rowidx  = (int*)alloc(B_ * sizeof(int));
  uint16_t* xb   = (uint16_t*)alloc((size_t)B_ * DP * 2);
  uint16_t* w1t  = (uint16_t*)alloc((size_t)T_ * H_ * DP * 2);        // [T][H][DP]
  uint16_t* w2t  = (uint16_t*)alloc((size_t)T_ * H_ * H_ * 2);
  uint16_t* w3t  = (uint16_t*)alloc((size_t)T_ * H_ * H_ * 2);
  uint16_t* w4t  = (uint16_t*)alloc((size_t)T_ * (2 * L_) * H_ * 2);  // [T][128][256]
  uint16_t* s1t  = (uint16_t*)alloc((size_t)H_ * L_ * 2);             // [256][64]
  uint16_t* s2t  = (uint16_t*)alloc((size_t)H_ * H_ * 2);
  uint16_t* hw1t = (uint16_t*)alloc((size_t)T_ * H_ * H_ * 2);
  uint16_t* hw2t = (uint16_t*)alloc((size_t)T_ * D_ * H_ * 2);        // [T][784][256]
  float* s1b = (float*)alloc(H_ * 4);
  float* s2b = (float*)alloc(H_ * 4);
  float* hb1 = (float*)alloc((size_t)T_ * H_ * 4);
  float* hb2 = (float*)alloc((size_t)T_ * D_ * 4);
  uint16_t* act1 = (uint16_t*)alloc((size_t)B_ * H_ * 2);
  uint16_t* act2 = (uint16_t*)alloc((size_t)B_ * H_ * 2);
  float* encout  = (float*)alloc((size_t)B_ * (2 * L_) * 4);
  uint16_t* zb   = (uint16_t*)alloc((size_t)B_ * L_ * 2);

  // 1) bucket rows by task
  k_zero_i32<<<1, 32, 0, stream>>>(counts, T_);
  k_count<<<div_up(B_, 256), 256, 0, stream>>>(task, counts);
  k_scan<<<1, 32, 0, stream>>>(counts, offsets, cursors);
  k_scatter<<<div_up(B_, 256), 256, 0, stream>>>(task, cursors, rowidx);

  // 2) convert x + deterministic encoder weights to bf16 (transposed)
  k_cvt_x<<<div_up((long)B_ * DP, 256), 256, 0, stream>>>(x, xb);
  k_cvt_wt<<<div_up((long)T_ * H_ * DP, 256), 256, 0, stream>>>(enc_W1, w1t, T_, D_, H_, DP);
  k_cvt_wt<<<div_up((long)T_ * H_ * H_, 256), 256, 0, stream>>>(enc_W2, w2t, T_, H_, H_, H_);
  k_cvt_wt<<<div_up((long)T_ * H_ * H_, 256), 256, 0, stream>>>(enc_W3, w3t, T_, H_, H_, H_);
  k_cvt_wt<<<div_up((long)T_ * (2 * L_) * H_, 256), 256, 0, stream>>>(enc_W4, w4t, T_, H_, 2 * L_, H_);

  // 3) sample Bayesian weights (mu + exp(ls)*eps), transposed bf16
  k_sample_wt<<<div_up((long)H_ * L_, 256), 256, 0, stream>>>(s1_wmu, s1_wls, eps_s1_w, s1t, 1, L_, H_, L_);
  k_sample_wt<<<div_up((long)H_ * H_, 256), 256, 0, stream>>>(s2_wmu, s2_wls, eps_s2_w, s2t, 1, H_, H_, H_);
  k_sample_wt<<<div_up((long)T_ * H_ * H_, 256), 256, 0, stream>>>(h1_wmu, h1_wls, eps_h1_w, hw1t, T_, H_, H_, H_);
  k_sample_wt<<<div_up((long)T_ * D_ * H_, 256), 256, 0, stream>>>(h2_wmu, h2_wls, eps_h2_w, hw2t, T_, H_, D_, H_);
  k_sample_b<<<div_up(H_, 256), 256, 0, stream>>>(s1_bmu, s1_bls, eps_s1_b, s1b, H_);
  k_sample_b<<<div_up(H_, 256), 256, 0, stream>>>(s2_bmu, s2_bls, eps_s2_b, s2b, H_);
  k_sample_b<<<div_up(T_ * H_, 256), 256, 0, stream>>>(h1_bmu, h1_bls, eps_h1_b, hb1, T_ * H_);
  k_sample_b<<<div_up(T_ * D_, 256), 256, 0, stream>>>(h2_bmu, h2_bls, eps_h2_b, hb2, T_ * D_);

  dim3 gGather(T_ * MAXT_TILES, 1);
  dim3 gDense(B_ / MROWS, 1);

  // 4) encoder (gathered per-task tiles)
  k_gemm<<<gGather, 256, 0, stream>>>(xb,   w1t, enc_b1, nullptr, act1, rowidx, offsets, H_, DP, H_, 0);
  k_gemm<<<gGather, 256, 0, stream>>>(act1, w2t, enc_b2, nullptr, act2, rowidx, offsets, H_, H_, H_, 0);
  k_gemm<<<gGather, 256, 0, stream>>>(act2, w3t, enc_b3, nullptr, act1, rowidx, offsets, H_, H_, H_, 0);
  k_gemm<<<gGather, 256, 0, stream>>>(act1, w4t, enc_b4, encout, nullptr, rowidx, offsets, 2 * L_, H_, 2 * L_, 1);

  // 5) reparameterize + emit mu / log_sigma
  k_z<<<div_up((long)B_ * L_, 256), 256, 0, stream>>>(encout, eps_z, muO, lsO, zb);

  // 6) shared decoder (no task gather)
  k_gemm<<<gDense, 256, 0, stream>>>(zb,   s1t, s1b, nullptr, act1, nullptr, nullptr, H_, L_, H_, 0);
  k_gemm<<<gDense, 256, 0, stream>>>(act1, s2t, s2b, nullptr, act2, nullptr, nullptr, H_, H_, H_, 0);

  // 7) task heads (gathered); head2 writes sigmoid(gen) straight to d_out
  k_gemm<<<gGather, 256, 0, stream>>>(act2, hw1t, hb1, nullptr, act1, rowidx, offsets, H_, H_, H_, 0);
  dim3 gHead2(T_ * MAXT_TILES, div_up(div_up(D_, 32), 8));
  k_gemm<<<gHead2, 256, 0, stream>>>(act1, hw2t, hb2, genO, nullptr, rowidx, offsets, D_, H_, D_, 2);
}
